// CausalSelfAttention_20375324852479
// MI455X (gfx1250) — matrix operand order repaired and checked by reading, not compiled, not run on hardware
//
#include <hip/hip_runtime.h>
#include <hip/hip_bf16.h>
#include <stdint.h>

#define D_MODEL_  2048
#define N_HEADS_  16
#define KV_HEADS_ 4
#define HEAD_DIM_ 128
#define GROUP_    4
#define KV_WIDTH_ 512
#define BATCH_    2
#define SEQ_      2048
#define QKV_N_    3072      // D_MODEL + 2*KV_WIDTH
#define MROWS_    4096      // BATCH*SEQ

typedef __attribute__((ext_vector_type(16))) __bf16 v16bf;
typedef __attribute__((ext_vector_type(8)))  float  v8f;

struct Frag8 { unsigned int u[8]; };

static __device__ __forceinline__ v16bf frag_bf(const Frag8 &f) {
  return __builtin_bit_cast(v16bf, f);
}
static __device__ __forceinline__ unsigned short f2bf(float x) {
  unsigned int u = __builtin_bit_cast(unsigned int, x);
  u += 0x7FFFu + ((u >> 16) & 1u);          // round-to-nearest-even
  return (unsigned short)(u >> 16);
}

// ---------------------------------------------------------------- convert
__global__ __launch_bounds__(256)
void cvt_f32_bf16(const float* __restrict__ in, unsigned short* __restrict__ out, int n) {
  int i = blockIdx.x * blockDim.x + threadIdx.x;
  int stride = gridDim.x * blockDim.x;
  for (; i < n; i += stride) out[i] = f2bf(in[i]);
}

// ---------------------------------------------------------------- GEMM
// C[M,N] = A[M,K](bf16) @ B[K,N](bf16) + bias[N](f32)
// BM=128, BN=128, Kstep=32. 256 threads = 8 waves arranged 4(M) x 2(N);
// each wave computes 32 rows x 64 cols = 8 WMMA C tiles per K-step
// (A fragments reused over 4 N-tiles, B fragments over 2 M-tiles).
template<bool OUT_BF16>
__global__ __launch_bounds__(256)
void gemm_bf16_kernel(const unsigned short* __restrict__ A,
                      const unsigned short* __restrict__ Bw,
                      const float* __restrict__ bias,
                      void* __restrict__ C,
                      int M, int N, int K)
{
  __shared__ unsigned short As[128][32];   // [m][k] row-major
  __shared__ unsigned short Bs[128][32];   // transposed: [n][k]

  const int tid  = threadIdx.x;
  const int lane = tid & 31;
  const int w    = tid >> 5;
  const int mw   = w >> 1;        // 0..3  -> 32-row band
  const int nw   = w & 1;         // 0..1  -> 64-col band
  const int m0   = blockIdx.y * 128;
  const int n0   = blockIdx.x * 128;
  const int l16  = lane & 15;
  const int hi   = lane >> 4;

  v8f acc[2][4] = {};

  for (int k0 = 0; k0 < K; k0 += 32) {
    #pragma unroll
    for (int it = 0; it < 2; ++it) { // A tile: 128x32 bf16
      int seg = tid + it * 256;            // 0..511
      int row = seg >> 2, s = seg & 3;
      const unsigned short* gp = A + (size_t)(m0 + row) * K + k0 + s * 8;
      uint4 v = *(const uint4*)gp;
      *(uint4*)&As[row][s * 8] = v;
      if (k0 + 32 < K) __builtin_prefetch(gp + 32, 0, 3);
    }
    #pragma unroll
    for (int it = 0; it < 2; ++it) { // B tile: 32x128 bf16, stored transposed
      int seg = tid + it * 256;            // 0..511
      int kr = seg >> 4, s = seg & 15;
      const unsigned short* gp = Bw + (size_t)(k0 + kr) * N + n0 + s * 8;
      uint4 v = *(const uint4*)gp;
      const unsigned short* e = (const unsigned short*)&v;
      #pragma unroll
      for (int q = 0; q < 8; ++q) Bs[s * 8 + q][kr] = e[q];
      if (k0 + 32 < K) __builtin_prefetch(gp + (size_t)32 * N, 0, 3);
    }
    __syncthreads();

    Frag8 af[2];   // two A fragments (16x32 each): lane=M(l16), K pairs per VGPR
    #pragma unroll
    for (int mt = 0; mt < 2; ++mt)
      #pragma unroll
      for (int v = 0; v < 8; ++v) {
        int kk = ((v < 4) ? 0 : 16) + hi * 8 + (v & 3) * 2;
        af[mt].u[v] = *(const unsigned int*)&As[mw * 32 + mt * 16 + l16][kk];
      }
    #pragma unroll
    for (int nt = 0; nt < 4; ++nt) {
      Frag8 bfr; // B fragment 32x16: lane=N, same K pairs per VGPR as A
      #pragma unroll
      for (int v = 0; v < 8; ++v)
        bfr.u[v] = *(const unsigned int*)&Bs[nw * 64 + nt * 16 + l16][((v < 4) ? 0 : 16) + hi * 8 + (v & 3) * 2];
      #pragma unroll
      for (int mt = 0; mt < 2; ++mt)
        acc[mt][nt] = __builtin_amdgcn_wmma_f32_16x16x32_bf16(
            false, frag_bf(af[mt]), false, frag_bf(bfr), (short)0, acc[mt][nt],
            false, false);
    }
    __syncthreads();
  }

  // epilogue: + bias, store
  #pragma unroll
  for (int nt = 0; nt < 4; ++nt) {
    int n = n0 + nw * 64 + nt * 16 + l16;
    float bv = bias[n];
    #pragma unroll
    for (int mt = 0; mt < 2; ++mt)
      #pragma unroll
      for (int r = 0; r < 8; ++r) {
        int m = m0 + mw * 32 + mt * 16 + r + 8 * hi;
        float val = acc[mt][nt][r] + bv;
        if (OUT_BF16) ((unsigned short*)C)[(size_t)m * N + n] = f2bf(val);
        else          ((float*)C)[(size_t)m * N + n] = val;
      }
  }
}

// ---------------------------------------------------------------- attention
// Flash-style causal GQA. Block = 128 threads (4 waves); each wave owns a
// 16-query tile; key blocks of 32. grid = (SEQ/64, N_HEADS, BATCH).
__global__ __launch_bounds__(128)
void attn_kernel(const unsigned short* __restrict__ qkv,  // [MROWS][QKV_N] bf16
                 unsigned short* __restrict__ outb)       // [MROWS][D_MODEL] bf16
{
  __shared__ unsigned short Kt[32][128];       // [key][dim]
  __shared__ unsigned short Vt[128][32];       // transposed [dim][key]
  __shared__ float          Ssc[4][16][32];    // per-wave S scratch
  __shared__ unsigned short Pp[4][16][32];     // per-wave P (bf16)
  __shared__ float          rowA[4][16];       // per-row alpha
  __shared__ float          rowL[4][16];       // per-row l

  const int tid  = threadIdx.x;
  const int lane = tid & 31;
  const int w    = tid >> 5;
  const int l16  = lane & 15;
  const int hi   = lane >> 4;
  const int h    = blockIdx.y;
  const int kv   = h / GROUP_;
  const int b    = blockIdx.z;
  const int q0b  = blockIdx.x * 64;
  const int qrow0 = q0b + w * 16;

  const int qcol = h * HEAD_DIM_;
  const int kcol = D_MODEL_ + kv * HEAD_DIM_;
  const int vcol = D_MODEL_ + KV_WIDTH_ + kv * HEAD_DIM_;

  // Q fragments (4 K-chunks of 32), held in registers for the whole loop
  unsigned int qa[4][8];
  {
    const size_t rowbase = (size_t)(b * SEQ_ + qrow0 + l16) * QKV_N_ + qcol;
    #pragma unroll
    for (int c = 0; c < 4; ++c)
      #pragma unroll
      for (int v = 0; v < 8; ++v) {
        int kk = ((v < 4) ? 0 : 16) + hi * 8 + (v & 3) * 2;
        qa[c][v] = *(const unsigned int*)(qkv + rowbase + c * 32 + kk);
      }
  }

  v8f o[8] = {};
  float rmax = -1e30f, rsum = 0.0f;
  const int jmax = (q0b + 63) >> 5;   // inclusive key-block bound for this block
  const float scale = 0.08838834764831845f; // 1/sqrt(128)

  for (int j = 0; j <= jmax; ++j) {
    // cooperative load: K block 32x128 (row-major), V block transposed
    #pragma unroll
    for (int it = 0; it < 4; ++it) {
      int seg = tid + it * 128;           // 0..511
      int key = seg >> 4, s = seg & 15;
      size_t base = (size_t)(b * SEQ_ + j * 32 + key) * QKV_N_;
      const unsigned short* kp = qkv + base + kcol + s * 8;
      const unsigned short* vp = qkv + base + vcol + s * 8;
      uint4 kvv = *(const uint4*)kp;
      *(uint4*)&Kt[key][s * 8] = kvv;
      uint4 vvv = *(const uint4*)vp;
      const unsigned short* e = (const unsigned short*)&vvv;
      #pragma unroll
      for (int q = 0; q < 8; ++q) Vt[s * 8 + q][key] = e[q];
      if (j < jmax) {                    // prefetch next key block
        __builtin_prefetch(kp + (size_t)32 * QKV_N_, 0, 3);
        __builtin_prefetch(vp + (size_t)32 * QKV_N_, 0, 3);
      }
    }
    __syncthreads();

    if (j * 32 <= qrow0 + 15) {
      // ---- S = Q * K^T, two 16-key tiles
      #pragma unroll
      for (int st = 0; st < 2; ++st) {
        v8f sacc = {};
        #pragma unroll
        for (int c = 0; c < 4; ++c) {
          Frag8 aq, kb;
          #pragma unroll
          for (int v = 0; v < 8; ++v) aq.u[v] = qa[c][v];
          #pragma unroll
          for (int v = 0; v < 8; ++v)
            kb.u[v] = *(const unsigned int*)&Kt[st * 16 + l16][c * 32 + ((v < 4) ? 0 : 16) + hi * 8 + (v & 3) * 2];
          sacc = __builtin_amdgcn_wmma_f32_16x16x32_bf16(
              false, frag_bf(aq), false, frag_bf(kb), (short)0, sacc, false, false);
        }
        #pragma unroll
        for (int r = 0; r < 8; ++r) {        // scale + causal mask -> LDS
          int m  = r + 8 * hi;
          int nl = st * 16 + l16;
          int keyg = j * 32 + nl;
          float val = sacc[r] * scale;
          if (keyg > qrow0 + m) val = -1e30f;
          Ssc[w][m][nl] = val;
        }
      }
      __builtin_amdgcn_wave_barrier();

      // ---- online softmax: lane L (<16) owns row L
      if (lane < 16) {
        int m = lane;
        float mprev = rmax, mnew = mprev;
        #pragma unroll
        for (int n = 0; n < 32; ++n) mnew = fmaxf(mnew, Ssc[w][m][n]);
        float alpha = __expf(mprev - mnew);
        float ps = 0.0f;
        #pragma unroll
        for (int n = 0; n < 32; ++n) {
          float p = __expf(Ssc[w][m][n] - mnew);
          ps += p;
          Pp[w][m][n] = f2bf(p);
        }
        rsum = rsum * alpha + ps;
        rmax = mnew;
        rowA[w][m] = alpha;
      }
      __builtin_amdgcn_wave_barrier();

      // ---- O = O*alpha + P @ V
      Frag8 pf;
      #pragma unroll
      for (int v = 0; v < 8; ++v) {
        int kk = ((v < 4) ? 0 : 16) + hi * 8 + (v & 3) * 2;
        pf.u[v] = *(const unsigned int*)&Pp[w][l16][kk];
      }
      float al[8];
      #pragma unroll
      for (int r = 0; r < 8; ++r) al[r] = rowA[w][r + 8 * hi];
      #pragma unroll
      for (int nt = 0; nt < 8; ++nt) {
        v8f t = o[nt];
        #pragma unroll
        for (int r = 0; r < 8; ++r) t[r] *= al[r];
        Frag8 vf;
        #pragma unroll
        for (int v = 0; v < 8; ++v)
          vf.u[v] = *(const unsigned int*)&Vt[nt * 16 + l16][((v < 4) ? 0 : 16) + hi * 8 + (v & 3) * 2];
        o[nt] = __builtin_amdgcn_wmma_f32_16x16x32_bf16(
            false, frag_bf(pf), false, frag_bf(vf), (short)0, t, false, false);
      }
    }
    __syncthreads();
  }

  if (lane < 16) rowL[w][lane] = rsum;
  __builtin_amdgcn_wave_barrier();
  float inv[8];
  #pragma unroll
  for (int r = 0; r < 8; ++r) inv[r] = 1.0f / rowL[w][r + 8 * hi];
  #pragma unroll
  for (int nt = 0; nt < 8; ++nt)
    #pragma unroll
    for (int r = 0; r < 8; ++r) {
      int m = r + 8 * hi;
      int col = h * HEAD_DIM_ + nt * 16 + l16;
      outb[(size_t)(b * SEQ_ + qrow0 + m) * D_MODEL_ + col] = f2bf(o[nt][r] * inv[r]);
    }
}

// ---------------------------------------------------------------- launch
extern "C" void kernel_launch(void* const* d_in, const int* in_sizes, int n_in,
                              void* d_out, int out_size, void* d_ws, size_t ws_size,
                              hipStream_t stream) {
  (void)in_sizes; (void)n_in; (void)out_size; (void)ws_size;
  const float* x      = (const float*)d_in[0];
  /* d_in[1] = attn_mask: exactly tril, computed from indices instead */
  const float* qkv_w  = (const float*)d_in[2];
  const float* qkv_b  = (const float*)d_in[3];
  const float* proj_w = (const float*)d_in[4];
  const float* proj_b = (const float*)d_in[5];
  float* out = (float*)d_out;

  char* p = (char*)d_ws;
  unsigned short* xb    = (unsigned short*)p; p += (size_t)MROWS_ * D_MODEL_ * 2;   // 16.78 MB
  unsigned short* wqkv  = (unsigned short*)p; p += (size_t)D_MODEL_ * QKV_N_ * 2;   // 12.58 MB
  unsigned short* wproj = (unsigned short*)p; p += (size_t)D_MODEL_ * D_MODEL_ * 2; //  8.39 MB
  unsigned short* qkvb  = (unsigned short*)p; p += (size_t)MROWS_ * QKV_N_ * 2;     // 25.17 MB
  unsigned short* attnb = (unsigned short*)p;                                       // 16.78 MB

  // 1) convert fp32 -> bf16
  cvt_f32_bf16<<<4096, 256, 0, stream>>>(x,      xb,    MROWS_ * D_MODEL_);
  cvt_f32_bf16<<<4096, 256, 0, stream>>>(qkv_w,  wqkv,  D_MODEL_ * QKV_N_);
  cvt_f32_bf16<<<4096, 256, 0, stream>>>(proj_w, wproj, D_MODEL_ * D_MODEL_);

  // 2) qkv = x @ qkv_w + qkv_b  (bf16 out)
  dim3 g1(QKV_N_ / 128, MROWS_ / 128);
  gemm_bf16_kernel<true><<<g1, 256, 0, stream>>>(xb, wqkv, qkv_b, qkvb,
                                                 MROWS_, QKV_N_, D_MODEL_);

  // 3) causal GQA flash attention (bf16 out)
  dim3 ga(SEQ_ / 64, N_HEADS_, BATCH_);
  attn_kernel<<<ga, 128, 0, stream>>>(qkvb, attnb);

  // 4) out = attn @ proj_w + proj_b  (f32 out)
  dim3 g2(D_MODEL_ / 128, MROWS_ / 128);
  gemm_bf16_kernel<false><<<g2, 256, 0, stream>>>(attnb, wproj, proj_b, out,
                                                  MROWS_, D_MODEL_, D_MODEL_);
}
